// MixedAttention_16561393893612
// MI455X (gfx1250) — matrix operand order repaired and checked by reading, not compiled, not run on hardware
//
#include <hip/hip_runtime.h>

// ---------------- problem constants (from reference) ----------------
#define TT   2048      // tokens
#define DM   1024      // d_model
#define NH   16        // heads
#define DH   64        // head dim
#define NNS  64        // non-shared tokens
#define NS   (TT-NNS)  // 1984 shared tokens
#define BB   4         // batch

typedef __attribute__((ext_vector_type(16))) __bf16 v16bf;
typedef __attribute__((ext_vector_type(8)))  __bf16 v8bf;
typedef __attribute__((ext_vector_type(8)))  float  v8f;

__device__ __forceinline__ unsigned short f2bf_bits(float f) {
  unsigned u = __float_as_uint(f);
  unsigned r = u + 0x7FFFu + ((u >> 16) & 1u);   // round-to-nearest-even
  return (unsigned short)(r >> 16);
}

__device__ __forceinline__ v8f vzero8() {
  v8f z;
#pragma unroll
  for (int i = 0; i < 8; ++i) z[i] = 0.0f;
  return z;
}

__device__ __forceinline__ v16bf cat8(v8bf lo, v8bf hi) {
  v16bf a;
#pragma unroll
  for (int i = 0; i < 8; ++i) { a[i] = lo[i]; a[8 + i] = hi[i]; }
  return a;
}

__device__ __forceinline__ v8f wmma_bf16(v16bf a, v16bf b, v8f c) {
  // D = A(16x32 bf16) * B(32x16 bf16) + C(16x16 f32)
  return __builtin_amdgcn_wmma_f32_16x16x32_bf16(false, a, false, b, (short)0, c,
                                                 false, false);
}

// ---------------- f32 -> bf16 conversion ----------------
__global__ __launch_bounds__(256) void f32_to_bf16(const float* __restrict__ src,
                                                   unsigned short* __restrict__ dst,
                                                   int n) {
  int i = (blockIdx.x * 256 + threadIdx.x) * 4;
  if (i >= n) return;
  float4 v = *(const float4*)(src + i);
  unsigned lo = (unsigned)f2bf_bits(v.x) | ((unsigned)f2bf_bits(v.y) << 16);
  unsigned hi = (unsigned)f2bf_bits(v.z) | ((unsigned)f2bf_bits(v.w) << 16);
  uint2 pk; pk.x = lo; pk.y = hi;
  *(uint2*)(dst + i) = pk;
}

// ---------------- WMMA GEMM: C[M,N] = A_bf16[M,1024] * W_bf16[N,1024]^T ---------
// EPI==0: store f32 to Cout (row-major, ld = Ncols)         (output projection)
// EPI==1: scatter bf16 into Q[B,H,T,Dh], K[B,H,T,Dh], Vt[B,H,Dh,T]  (shared QKV)
template <int EPI>
__global__ __launch_bounds__(256) void gemm_bf16(
    const unsigned short* __restrict__ A, const unsigned short* __restrict__ W,
    float* __restrict__ Cout, unsigned short* __restrict__ Qb,
    unsigned short* __restrict__ Kb, unsigned short* __restrict__ Vtb, int Ncols) {
  const int lane = threadIdx.x & 31;
  const int wave = threadIdx.x >> 5;
  const int wm = wave & 3;          // 4 waves along M
  const int wn = wave >> 2;         // 2 waves along N
  const int mBase = blockIdx.y * 128 + wm * 32;   // 32 rows per wave
  const int nBase = blockIdx.x * 128 + wn * 64;   // 64 cols per wave
  const int half = lane >> 4;
  const int l16 = lane & 15;

  // per-lane A row pointers (A fragment: M = lane%16)
  const unsigned short* arow[2];
#pragma unroll
  for (int mt = 0; mt < 2; ++mt) {
    int grow = mBase + mt * 16 + l16;
    long long addr;
    if (EPI == 1) {                 // rows are (b, t<NS) tokens of x
      int b = grow / NS;
      int t = grow - b * NS;
      addr = (long long)(b * TT + t) * DM;
    } else {
      addr = (long long)grow * DM;
    }
    arow[mt] = A + addr;
  }
  // per-lane B row pointer (B fragment: N = lane%16; W row n is contiguous in K)
  const unsigned short* brow = W + (long long)(nBase + l16) * DM;

  v8f acc[2][4];
#pragma unroll
  for (int mt = 0; mt < 2; ++mt)
#pragma unroll
    for (int nt = 0; nt < 4; ++nt) acc[mt][nt] = vzero8();

  for (int kk = 0; kk < DM; kk += 32) {
    __builtin_prefetch(brow + kk + 64, 0, 0);     // global_prefetch_b8
    v16bf af[2];
#pragma unroll
    for (int mt = 0; mt < 2; ++mt) {
      v8bf lo = *(const v8bf*)(arow[mt] + kk + half * 8);        // K 0..7 / 8..15
      v8bf hi = *(const v8bf*)(arow[mt] + kk + 16 + half * 8);   // K 16..23 / 24..31
      af[mt] = cat8(lo, hi);
    }
#pragma unroll
    for (int nt = 0; nt < 4; ++nt) {
      v16bf bf = cat8(*(const v8bf*)(brow + (long long)nt * 16 * DM + kk + half * 8), *(const v8bf*)(brow + (long long)nt * 16 * DM + kk + 16 + half * 8));
#pragma unroll
      for (int mt = 0; mt < 2; ++mt) acc[mt][nt] = wmma_bf16(af[mt], bf, acc[mt][nt]);
    }
  }

#pragma unroll
  for (int nt = 0; nt < 4; ++nt) {
    const int col = nBase + nt * 16 + l16;
    const int p = col >> 10, rem = col & 1023;
    const int h = rem >> 6, d = rem & 63;
#pragma unroll
    for (int mt = 0; mt < 2; ++mt) {
#pragma unroll
      for (int r = 0; r < 8; ++r) {
        const int row = mBase + mt * 16 + r + half * 8;   // C layout: M = r + 8*half
        if (EPI == 0) {
          Cout[(long long)row * Ncols + col] = acc[mt][nt][r];
        } else {
          int b = row / NS;
          int t = row - b * NS;
          unsigned short v = f2bf_bits(acc[mt][nt][r]);
          int bh = b * NH + h;
          if (p == 0)      Qb[((long long)bh * TT + t) * DH + d] = v;
          else if (p == 1) Kb[((long long)bh * TT + t) * DH + d] = v;
          else             Vtb[((long long)bh * DH + d) * TT + t] = v;
        }
      }
    }
  }
}

// ---------------- ns-token projection: pure HBM-bound f32 GEMV ----------------
// out[b, NS+t, m] = dot(x[b,NS+t,:], W_ns[t,m,:]) for 4 batches at once.
__global__ __launch_bounds__(256) void ns_proj(const float* __restrict__ x,
                                               const float* __restrict__ Wns,
                                               unsigned short* __restrict__ Qb,
                                               unsigned short* __restrict__ Kb,
                                               unsigned short* __restrict__ Vtb) {
  const int lane = threadIdx.x & 31;
  const int wave = threadIdx.x >> 5;
  const int pair = blockIdx.x * 8 + wave;          // (t,m) pair, t*3072+m
  const int t = pair / (3 * DM);
  const int m = pair - t * (3 * DM);
  const float4* wp = (const float4*)(Wns + (long long)pair * DM);
  const float4* x0 = (const float4*)(x + (long long)(0 * TT + NS + t) * DM);
  const float4* x1 = (const float4*)(x + (long long)(1 * TT + NS + t) * DM);
  const float4* x2 = (const float4*)(x + (long long)(2 * TT + NS + t) * DM);
  const float4* x3 = (const float4*)(x + (long long)(3 * TT + NS + t) * DM);
  float a0 = 0.f, a1 = 0.f, a2 = 0.f, a3 = 0.f;
#pragma unroll
  for (int j = 0; j < 8; ++j) {
    float4 w = wp[lane + j * 32];
    float4 v;
    v = x0[lane + j * 32]; a0 += w.x * v.x + w.y * v.y + w.z * v.z + w.w * v.w;
    v = x1[lane + j * 32]; a1 += w.x * v.x + w.y * v.y + w.z * v.z + w.w * v.w;
    v = x2[lane + j * 32]; a2 += w.x * v.x + w.y * v.y + w.z * v.z + w.w * v.w;
    v = x3[lane + j * 32]; a3 += w.x * v.x + w.y * v.y + w.z * v.z + w.w * v.w;
  }
#pragma unroll
  for (int mask = 1; mask < 32; mask <<= 1) {
    a0 += __shfl_xor(a0, mask, 32);
    a1 += __shfl_xor(a1, mask, 32);
    a2 += __shfl_xor(a2, mask, 32);
    a3 += __shfl_xor(a3, mask, 32);
  }
  if (lane < 4) {
    const int p = m >> 10, rem = m & 1023, h = rem >> 6, d = rem & 63;
    float v = (lane == 0) ? a0 : (lane == 1) ? a1 : (lane == 2) ? a2 : a3;
    unsigned short bv = f2bf_bits(v);
    int bh = lane * NH + h;
    int tok = NS + t;
    if (p == 0)      Qb[((long long)bh * TT + tok) * DH + d] = bv;
    else if (p == 1) Kb[((long long)bh * TT + tok) * DH + d] = bv;
    else             Vtb[((long long)bh * DH + d) * TT + tok] = bv;
  }
}

// ---------------- flash attention (wave32 WMMA, online softmax) ----------------
__global__ __launch_bounds__(256) void flash_attn(const unsigned short* __restrict__ Qb,
                                                  const unsigned short* __restrict__ Kb,
                                                  const unsigned short* __restrict__ Vtb,
                                                  unsigned short* __restrict__ ctx) {
  __shared__ __align__(16) unsigned short ldsP[8 * 16 * 64];   // per-wave 16x64 P tile
  const int lane = threadIdx.x & 31;
  const int wave = threadIdx.x >> 5;
  const int half = lane >> 4;
  const int l16 = lane & 15;
  const int bh = blockIdx.y;             // b*16 + h
  const int b = bh >> 4, h = bh & 15;
  const int qBase = blockIdx.x * 128 + wave * 16;

  // Q A-fragments (K = Dh = 64 -> two k-steps of 32)
  const unsigned short* qrow = Qb + ((long long)bh * TT + qBase + l16) * DH;
  v16bf aq[2];
#pragma unroll
  for (int s = 0; s < 2; ++s) {
    v8bf lo = *(const v8bf*)(qrow + s * 32 + half * 8);
    v8bf hi = *(const v8bf*)(qrow + s * 32 + 16 + half * 8);
    aq[s] = cat8(lo, hi);
  }

  v8f accO[4];
#pragma unroll
  for (int j = 0; j < 4; ++j) accO[j] = vzero8();
  float mrow[8], lrow[8];
#pragma unroll
  for (int r = 0; r < 8; ++r) { mrow[r] = -__builtin_inff(); lrow[r] = 0.f; }

  unsigned short* myP = ldsP + wave * 16 * 64;
  const float sc = 0.125f;   // 1/sqrt(64)

  for (int kb = 0; kb < TT; kb += 64) {
    // S = Q * K^T  (4 key tiles of 16)
    v8f s[4];
#pragma unroll
    for (int j = 0; j < 4; ++j) {
      const unsigned short* kcol = Kb + ((long long)bh * TT + kb + j * 16 + l16) * DH;
      v16bf b0 = cat8(*(const v8bf*)(kcol + half * 8), *(const v8bf*)(kcol + 16 + half * 8));         // d 0..31
      v16bf b1 = cat8(*(const v8bf*)(kcol + 32 + half * 8), *(const v8bf*)(kcol + 48 + half * 8));    // d 32..63
      v8f c = wmma_bf16(aq[0], b0, vzero8());
      s[j] = wmma_bf16(aq[1], b1, c);
    }
    // online softmax over the 64-column key block
#pragma unroll
    for (int r = 0; r < 8; ++r) {
      float mx = s[0][r] * sc;
#pragma unroll
      for (int j = 1; j < 4; ++j) mx = fmaxf(mx, s[j][r] * sc);
#pragma unroll
      for (int mask = 1; mask < 16; mask <<= 1) mx = fmaxf(mx, __shfl_xor(mx, mask, 16));
      float mnew = fmaxf(mrow[r], mx);
      float corr = __expf(mrow[r] - mnew);
      mrow[r] = mnew;
      float psum = 0.f;
#pragma unroll
      for (int j = 0; j < 4; ++j) {
        float p = __expf(s[j][r] * sc - mnew);
        s[j][r] = p;
        psum += p;
      }
#pragma unroll
      for (int mask = 1; mask < 16; mask <<= 1) psum += __shfl_xor(psum, mask, 16);
      lrow[r] = lrow[r] * corr + psum;
#pragma unroll
      for (int j = 0; j < 4; ++j) accO[j][r] *= corr;
    }
    // P (C layout) -> LDS -> A-fragment layout
#pragma unroll
    for (int j = 0; j < 4; ++j)
#pragma unroll
      for (int r = 0; r < 8; ++r)
        myP[(r + half * 8) * 64 + j * 16 + l16] = f2bf_bits(s[j][r]);
    __syncthreads();
    v16bf pa[2];
#pragma unroll
    for (int s2 = 0; s2 < 2; ++s2) {
      v8bf lo = *(const v8bf*)(myP + l16 * 64 + s2 * 32 + half * 8);
      v8bf hi = *(const v8bf*)(myP + l16 * 64 + s2 * 32 + 16 + half * 8);
      pa[s2] = cat8(lo, hi);
    }
    // O += P * V   (V^T layout -> contiguous B-fragment loads)
#pragma unroll
    for (int jd = 0; jd < 4; ++jd) {
      const unsigned short* vcol = Vtb + ((long long)bh * DH + jd * 16 + l16) * TT + kb;
      v16bf v0 = cat8(*(const v8bf*)(vcol + half * 8), *(const v8bf*)(vcol + 16 + half * 8));
      v16bf v1 = cat8(*(const v8bf*)(vcol + 32 + half * 8), *(const v8bf*)(vcol + 48 + half * 8));
      accO[jd] = wmma_bf16(pa[0], v0, accO[jd]);
      accO[jd] = wmma_bf16(pa[1], v1, accO[jd]);
    }
  }

  // normalize and store context [B*T, D] bf16
#pragma unroll
  for (int r = 0; r < 8; ++r) {
    float inv = 1.0f / lrow[r];
    int q = qBase + r + half * 8;
    unsigned short* orow = ctx + (long long)(b * TT + q) * DM + h * DH;
#pragma unroll
    for (int jd = 0; jd < 4; ++jd)
      orow[jd * 16 + l16] = f2bf_bits(accO[jd][r] * inv);
  }
}

// ---------------- host-side launch ----------------
extern "C" void kernel_launch(void* const* d_in, const int* in_sizes, int n_in,
                              void* d_out, int out_size, void* d_ws, size_t ws_size,
                              hipStream_t stream) {
  const float* x    = (const float*)d_in[0];
  // d_in[1] is n_s (== NS, known statically from the reference)
  const float* Ws   = (const float*)d_in[2];
  const float* Wns  = (const float*)d_in[3];
  const float* Wout = (const float*)d_in[4];
  float* out = (float*)d_out;

  char* ws = (char*)d_ws;
  size_t off = 0;
  auto alloc = [&](size_t bytes) -> void* {
    void* p = ws + off;
    off += (bytes + 255) & ~(size_t)255;
    return p;
  };
  unsigned short* x_bf    = (unsigned short*)alloc((size_t)BB * TT * DM * 2);
  unsigned short* Ws_bf   = (unsigned short*)alloc((size_t)3 * DM * DM * 2);
  unsigned short* Wout_bf = (unsigned short*)alloc((size_t)DM * DM * 2);
  unsigned short* Qb      = (unsigned short*)alloc((size_t)BB * NH * TT * DH * 2);
  unsigned short* Kb      = (unsigned short*)alloc((size_t)BB * NH * TT * DH * 2);
  unsigned short* Vtb     = (unsigned short*)alloc((size_t)BB * NH * DH * TT * 2);
  unsigned short* ctxb    = (unsigned short*)alloc((size_t)BB * TT * DM * 2);
  (void)ws_size; (void)in_sizes; (void)n_in; (void)out_size;

  // 1) conversions
  {
    int n = BB * TT * DM;
    f32_to_bf16<<<n / 4 / 256, 256, 0, stream>>>(x, x_bf, n);
    n = 3 * DM * DM;
    f32_to_bf16<<<n / 4 / 256, 256, 0, stream>>>(Ws, Ws_bf, n);
    n = DM * DM;
    f32_to_bf16<<<n / 4 / 256, 256, 0, stream>>>(Wout, Wout_bf, n);
  }
  // 2) shared-token QKV projection (WMMA), scatter to Q/K/Vt
  gemm_bf16<1><<<dim3(3 * DM / 128, (BB * NS) / 128), 256, 0, stream>>>(
      x_bf, Ws_bf, nullptr, Qb, Kb, Vtb, 3 * DM);
  // 3) non-shared-token projection (HBM-bound f32)
  ns_proj<<<(NNS * 3 * DM) / 8, 256, 0, stream>>>(x, Wns, Qb, Kb, Vtb);
  // 4) flash attention
  flash_attn<<<dim3(TT / 128, BB * NH), 256, 0, stream>>>(Qb, Kb, Vtb, ctxb);
  // 5) output projection (WMMA), f32 out
  gemm_bf16<0><<<dim3(DM / 128, (BB * TT) / 128), 256, 0, stream>>>(
      ctxb, Wout_bf, out, nullptr, nullptr, nullptr, DM);
}
